// FlaxBertSelfAttention_7301444403537
// MI455X (gfx1250) — matrix operand order repaired and checked by reading, not compiled, not run on hardware
//
#include <hip/hip_runtime.h>

typedef __attribute__((ext_vector_type(16))) __bf16 v16bf;
typedef __attribute__((ext_vector_type(8)))  __bf16 v8bf;
typedef __attribute__((ext_vector_type(4)))  __bf16 v4bf;
typedef __attribute__((ext_vector_type(8)))  float  v8f;

#define HIDDEN 1024
#define NHEADS 16
#define HDIM   64
#define BATCH  4
#define SEQ    1024
#define ROWS   (BATCH * SEQ)   /* 4096 */
#define NCOLS  (3 * HIDDEN)    /* 3072 */

#define LDS_STRIDE 40          /* 32 bf16 payload + 8 pad; 80B rows: 16B aligned, bank-spread */

// ---------------------------------------------------------------------------
// CDNA5 async copy: global -> LDS, 16 bytes per lane, tracked by ASYNCcnt.
// ---------------------------------------------------------------------------
__device__ inline void async_copy_b128(uint32_t lds_off, const __bf16* gptr) {
    asm volatile("global_load_async_to_lds_b128 %0, %1, off"
                 :: "v"(lds_off), "v"(gptr) : "memory");
}
#define WAIT_ASYNC(n) asm volatile("s_wait_asynccnt " #n ::: "memory")

// ---------------------------------------------------------------------------
// WMMA fragment loaders (wave32, bf16 16x16x32; contiguous k = 16*(lane/16) + s in both A and B)
// ---------------------------------------------------------------------------

// A-fragment (16x32) from row-major [M x ld] source (global or LDS).
// lane<16: row=lane, K [0..15]; lane>=16: same row, K [16..31]
__device__ inline v16bf load_a_bf16(const __bf16* base, int ld, int m0, int k0, int lane) {
    const __bf16* p = base + (size_t)(m0 + (lane & 15)) * ld + k0 + ((lane >> 4) << 4);
    v8bf lo = *(const v8bf*)p;
    v8bf hi = *(const v8bf*)(p + 8);
    v16bf r;
#pragma unroll
    for (int i = 0; i < 8; ++i) { r[i] = lo[i]; r[i + 8] = hi[i]; }
    return r;
}

// B-fragment (32x16) from N-major [N x ld] source, single 32B load (32B-aligned sources).
__device__ inline v16bf load_bt_bf16(const __bf16* base, int ld, int n0, int k0, int lane) {
    const __bf16* p = base + (size_t)(n0 + (lane & 15)) * ld + k0 + ((lane >> 4) << 4);
    return *(const v16bf*)p;
}

// B-fragment from N-major LDS (16B-aligned rows only): two 16B halves.
__device__ inline v16bf load_bt_lds(const __bf16* base, int ld, int n0, int k0, int lane) {
    const __bf16* p = base + (n0 + (lane & 15)) * ld + k0 + ((lane >> 4) << 4);
    v8bf lo = *(const v8bf*)p;
    v8bf hi = *(const v8bf*)(p + 8);
    v16bf r;
#pragma unroll
    for (int i = 0; i < 8; ++i) { r[i] = lo[i]; r[i + 8] = hi[i]; }
    return r;
}

// A-fragment from row-major f32 source (LDS probabilities), convert to bf16.
__device__ inline v16bf load_a_f32(const float* base, int ld, int m0, int k0, int lane) {
    const float* p = base + (m0 + (lane & 15)) * ld + k0 + ((lane >> 4) << 4);
    v16bf r;
#pragma unroll
    for (int i = 0; i < 8; ++i) { r[i] = (__bf16)p[i]; r[i + 8] = (__bf16)p[i + 8]; }
    return r;
}

// ---------------------------------------------------------------------------
// Kernel 1a: hidden f32 -> bf16, vectorized (float4 in, 4x bf16 out)
// ---------------------------------------------------------------------------
__global__ void cvt_a_kernel(const float* __restrict__ h, __bf16* __restrict__ a_bf) {
    int i = (blockIdx.x * 256 + threadIdx.x) * 4;
    float4 f = *(const float4*)(h + i);
    v4bf o;
    o[0] = (__bf16)f.x; o[1] = (__bf16)f.y; o[2] = (__bf16)f.z; o[3] = (__bf16)f.w;
    *(v4bf*)(a_bf + i) = o;
}

// ---------------------------------------------------------------------------
// Kernel 1b: LDS-tiled transpose: w(1024x3072) f32 -> wt(3072x1024) bf16.
// 32x32 tile per block; coalesced reads and coalesced bf16 writes.
// ---------------------------------------------------------------------------
__global__ void transpose_w_kernel(const float* __restrict__ w, __bf16* __restrict__ wt) {
    __shared__ float tile[32][33];
    int k0 = blockIdx.y * 32;          // 0..1023 (rows of w)
    int c0 = blockIdx.x * 32;          // 0..3071 (cols of w)
    int t = threadIdx.x;               // 256 threads
#pragma unroll
    for (int q = 0; q < 4; ++q) {
        int idx = t + 256 * q;         // 0..1023
        int kr = idx >> 5, cc = idx & 31;
        tile[kr][cc] = w[(size_t)(k0 + kr) * NCOLS + c0 + cc];
    }
    __syncthreads();
#pragma unroll
    for (int q = 0; q < 4; ++q) {
        int idx = t + 256 * q;
        int cc = idx >> 5, kr = idx & 31;
        wt[(size_t)(c0 + cc) * HIDDEN + k0 + kr] = (__bf16)tile[kr][cc];
    }
}

// ---------------------------------------------------------------------------
// Kernel 2: QKV GEMM, LDS-staged with CDNA5 async copies.
// Block = 256 threads (8 waves), block tile 128(M) x 128(N), K chunks of 32.
// Waves arranged 4(M) x 2(N); each wave computes 32x64 (2x4 16x16 tiles).
// Double-buffered LDS: A[128x40] + B[128x40] bf16 per buffer (40 KB total).
// Epilogue scatters into Q (scaled 1/8), K, Vt per the (b,s,1024,3) split.
// ---------------------------------------------------------------------------
__global__ void qkv_kernel(const __bf16* __restrict__ a_bf, const __bf16* __restrict__ wt,
                           const float* __restrict__ bias,
                           __bf16* __restrict__ qbuf, __bf16* __restrict__ kbuf,
                           __bf16* __restrict__ vtbuf) {
    __shared__ __bf16 smem[4 * 128 * LDS_STRIDE];   // A0,B0,A1,B1

    const int t    = threadIdx.x;
    const int lane = t & 31;
    const int wave = t >> 5;
    const int wm   = wave >> 1;        // 0..3 -> M offset wm*32
    const int wn   = wave & 1;         // 0..1 -> N offset wn*64
    const int n0   = blockIdx.x * 128;
    const int m0   = blockIdx.y * 128;

    const uint32_t smem_base = (uint32_t)(uintptr_t)(&smem[0]);
    const int BUF = 128 * LDS_STRIDE;  // elements per (A or B) tile: 5120

    // stage K-chunk kc into buffer `buf` (A: 128x32 from a_bf, B: 128x32 from wt)
    auto stage = [&](int kc, int buf) {
#pragma unroll
        for (int j = 0; j < 2; ++j) {
            int seg = t + 256 * j;     // 0..511
            int row = seg >> 2;        // 0..127
            int q   = seg & 3;         // 16B quarter (8 bf16)
            uint32_t loff = (uint32_t)(buf * 2 * BUF + row * LDS_STRIDE + q * 8) * 2;
            async_copy_b128(smem_base + loff,
                            a_bf + (size_t)(m0 + row) * HIDDEN + kc * 32 + q * 8);
            async_copy_b128(smem_base + loff + (uint32_t)BUF * 2,
                            wt + (size_t)(n0 + row) * HIDDEN + kc * 32 + q * 8);
        }
    };

    v8f acc[2][4] = {};
    stage(0, 0);
    for (int kc = 0; kc < HIDDEN / 32; ++kc) {
        int cur = kc & 1;
        if (kc + 1 < HIDDEN / 32) {
            stage(kc + 1, cur ^ 1);
            WAIT_ASYNC(0x4);           // our 4 stage-kc ops done; stage kc+1 in flight
        } else {
            WAIT_ASYNC(0x0);
        }
        __syncthreads();               // all waves' slices visible

        const __bf16* ab = smem + cur * 2 * BUF;
        const __bf16* bb = ab + BUF;
        v16bf af[2], bf[4];
#pragma unroll
        for (int i = 0; i < 2; ++i) af[i] = load_a_bf16(ab, LDS_STRIDE, wm * 32 + i * 16, 0, lane);
#pragma unroll
        for (int j = 0; j < 4; ++j) bf[j] = load_bt_lds(bb, LDS_STRIDE, wn * 64 + j * 16, 0, lane);
#pragma unroll
        for (int i = 0; i < 2; ++i)
#pragma unroll
            for (int j = 0; j < 4; ++j)
                acc[i][j] = __builtin_amdgcn_wmma_f32_16x16x32_bf16(
                    false, af[i], false, bf[j], (short)0, acc[i][j], false, false);
        __syncthreads();               // done reading before this buffer is restaged
    }

    // epilogue: C layout elem e, lane<16 -> (M=e, N=lane); lane>=16 -> (M=e+8, N=lane-16)
    const int N = lane & 15;
    const int mofs = (lane >> 4) << 3;
#pragma unroll
    for (int i = 0; i < 2; ++i) {
#pragma unroll
        for (int j = 0; j < 4; ++j) {
            int c = n0 + wn * 64 + j * 16 + N;
            int tt = c % 3;            // 0=q, 1=v, 2=k  (reshape (b,s,1024,3))
            int feat = c / 3;
            int head = feat >> 6;
            int d = feat & 63;
            float bv = bias[c];
#pragma unroll
            for (int e = 0; e < 8; ++e) {
                int r = m0 + wm * 32 + i * 16 + e + mofs;
                int bb2 = r >> 10;
                int s = r & 1023;
                float val = acc[i][j][e] + bv;
                size_t bh = (size_t)(bb2 * NHEADS + head);
                if (tt == 0) {
                    qbuf[(bh * SEQ + s) * HDIM + d] = (__bf16)(val * 0.125f);
                } else if (tt == 1) {
                    vtbuf[(bh * HDIM + d) * SEQ + s] = (__bf16)val;
                } else {
                    kbuf[(bh * SEQ + s) * HDIM + d] = (__bf16)val;
                }
            }
        }
    }
}

// ---------------------------------------------------------------------------
// Kernel 3: fused attention for one (batch, head, 16-query block).
// scores(16x1024) = Q @ K^T via WMMA -> LDS f32; masked softmax in LDS;
// out(16x64) = P @ V via WMMA with on-the-fly f32->bf16 A-fragments.
// Block = 128 threads (4 waves).
// ---------------------------------------------------------------------------
__global__ void attn_kernel(const __bf16* __restrict__ qbuf, const __bf16* __restrict__ kbuf,
                            const __bf16* __restrict__ vtbuf, const int* __restrict__ am,
                            float* __restrict__ out) {
    __shared__ float sc[16 * SEQ];     // 64 KB
    __shared__ float red[16][8];
    __shared__ float rowstat[16];

    int lane = threadIdx.x & 31;
    int wave = threadIdx.x >> 5;
    int q0   = blockIdx.x * 16;
    int head = blockIdx.y;
    int b    = blockIdx.z;
    size_t bh = (size_t)(b * NHEADS + head);
    const __bf16* Q  = qbuf  + bh * SEQ * HDIM;
    const __bf16* K  = kbuf  + bh * SEQ * HDIM;   // [s][d]: N-major B for Q@K^T
    const __bf16* Vt = vtbuf + bh * HDIM * SEQ;   // [d][s]: N-major B for P@V

    v16bf qf0 = load_a_bf16(Q, HDIM, q0, 0, lane);
    v16bf qf1 = load_a_bf16(Q, HDIM, q0, 32, lane);

    for (int kt = wave * 16; kt < wave * 16 + 16; ++kt) {
        int n0 = kt * 16;
        v8f acc = {};
        v16bf b0 = load_bt_bf16(K, HDIM, n0, 0, lane);
        acc = __builtin_amdgcn_wmma_f32_16x16x32_bf16(false, qf0, false, b0,
                                                      (short)0, acc, false, false);
        v16bf b1 = load_bt_bf16(K, HDIM, n0, 32, lane);
        acc = __builtin_amdgcn_wmma_f32_16x16x32_bf16(false, qf1, false, b1,
                                                      (short)0, acc, false, false);
        int N = lane & 15;
        int mofs = (lane >> 4) << 3;
#pragma unroll
        for (int i = 0; i < 8; ++i)
            sc[(i + mofs) * SEQ + n0 + N] = acc[i];
    }
    __syncthreads();

    // masked softmax: 16 rows x (8 threads x 128 cols)
    int row = threadIdx.x >> 3;
    int chunk = threadIdx.x & 7;
    int cbase = chunk * 128;
    float mx = -3.4e38f;
    for (int j = 0; j < 128; ++j) {
        int c = cbase + j;
        float mb = (am[b * SEQ + c] > 0) ? 0.0f : -1e10f;
        float v = sc[row * SEQ + c] + mb;
        sc[row * SEQ + c] = v;
        mx = fmaxf(mx, v);
    }
    red[row][chunk] = mx;
    __syncthreads();
    if (chunk == 0) {
        float m2 = red[row][0];
        for (int j = 1; j < 8; ++j) m2 = fmaxf(m2, red[row][j]);
        rowstat[row] = m2;
    }
    __syncthreads();
    float rmax = rowstat[row];
    float sum = 0.0f;
    for (int j = 0; j < 128; ++j) {
        int c = cbase + j;
        float e = __expf(sc[row * SEQ + c] - rmax);
        sc[row * SEQ + c] = e;
        sum += e;
    }
    red[row][chunk] = sum;
    __syncthreads();
    if (chunk == 0) {
        float s2 = 0.0f;
        for (int j = 0; j < 8; ++j) s2 += red[row][j];
        rowstat[row] = s2;
    }
    __syncthreads();
    float inv = 1.0f / rowstat[row];
    for (int j = 0; j < 128; ++j)
        sc[row * SEQ + cbase + j] *= inv;
    __syncthreads();

    // P @ V: each wave computes one 16x16 output tile (d-block = wave)
    int n0 = wave * 16;
    v8f acc = {};
    for (int k0 = 0; k0 < SEQ; k0 += 32) {
        v16bf a = load_a_f32(sc, SEQ, 0, k0, lane);
        v16bf bb = load_bt_bf16(Vt, SEQ, n0, k0, lane);
        acc = __builtin_amdgcn_wmma_f32_16x16x32_bf16(false, a, false, bb,
                                                      (short)0, acc, false, false);
    }
    int N = lane & 15;
    int mofs = (lane >> 4) << 3;
#pragma unroll
    for (int i = 0; i < 8; ++i) {
        int qrow = q0 + i + mofs;
        out[((size_t)(b * SEQ + qrow)) * HIDDEN + head * HDIM + n0 + N] = acc[i];
    }
}

// ---------------------------------------------------------------------------
extern "C" void kernel_launch(void* const* d_in, const int* in_sizes, int n_in,
                              void* d_out, int out_size, void* d_ws, size_t ws_size,
                              hipStream_t stream) {
    const float* hidden = (const float*)d_in[0];
    const int*   am     = (const int*)d_in[1];
    const float* w      = (const float*)d_in[2];
    const float* bias   = (const float*)d_in[3];
    float* out = (float*)d_out;

    char* ws = (char*)d_ws;
    __bf16* a_bf = (__bf16*)(ws);                         // 4096x1024 bf16 : 8 MB
    __bf16* wt   = (__bf16*)(ws + (size_t)8  * 1048576);  // 3072x1024 bf16 : 6 MB
    __bf16* qb   = (__bf16*)(ws + (size_t)14 * 1048576);  // [b][h][s][d]   : 8 MB
    __bf16* kb   = (__bf16*)(ws + (size_t)22 * 1048576);  // [b][h][s][d]   : 8 MB
    __bf16* vt   = (__bf16*)(ws + (size_t)30 * 1048576);  // [b][h][d][s]   : 8 MB

    // 1) activations f32->bf16 (vectorized) + LDS-tiled weight transpose
    cvt_a_kernel<<<(ROWS * HIDDEN) / (256 * 4), 256, 0, stream>>>(hidden, a_bf);
    {
        dim3 grid(NCOLS / 32, HIDDEN / 32);               // (96, 32)
        transpose_w_kernel<<<grid, 256, 0, stream>>>(w, wt);
    }
    // 2) QKV projection GEMM (async LDS staging) + scatter
    {
        dim3 grid(NCOLS / 128, ROWS / 128);               // (24, 32)
        qkv_kernel<<<grid, 256, 0, stream>>>(a_bf, wt, bias, qb, kb, vt);
    }
    // 3) fused attention
    {
        dim3 grid(SEQ / 16, NHEADS, BATCH);               // (64, 16, 4)
        attn_kernel<<<grid, 128, 0, stream>>>(qb, kb, vt, am, out);
    }
}
